// PLELayer_8263517077676
// MI455X (gfx1250) — compile-verified, matrix operand order checked by reading, not run on hardware
//
#include <hip/hip_runtime.h>
#include <hip/hip_bf16.h>

// ---------------------------------------------------------------------------
// PLE layer for MI455X (gfx1250, wave32, WMMA).
// T=3 tasks, B=16384, I=1024, O=512, 4 shared + 2 task experts, 6 gates.
//
// Pipeline (all on `stream`):
//   1) fp32 -> bf16 convert of x / shared_W / task_W into workspace
//   2) gate kernel: g[t][b][e] = softmax_e(x_t . gate_W_t + gate_b_t)
//   3) shared-expert GEMM pass: S[e][b][o] = relu(x0 . Ws_e + bs_e)  (bf16)
//   4) per-task fused GEMM: 2 task-expert WMMA GEMMs + bias + relu,
//      gated combine with the 4 shared-expert outputs from S.
// ---------------------------------------------------------------------------

typedef __attribute__((ext_vector_type(16))) __bf16 v16bf;
typedef __attribute__((ext_vector_type(8)))  __bf16 v8bf;
typedef __attribute__((ext_vector_type(8)))  float  v8f;
typedef __attribute__((ext_vector_type(8)))  unsigned short us8;

union U8 { us8 v; unsigned short s[8]; };

#define NT 3
#define NB 16384
#define NI 1024
#define NO 512
#define NE 6           // 4 shared + 2 task
#define BM 128
#define BN 128
#define KB 64
#define KBP 72         // padded LDS K stride (keeps 16B alignment: 144B rows)

static __device__ __forceinline__ unsigned short f2bf(float f) {
    unsigned u = __float_as_uint(f);
    unsigned r = u + 0x7FFFu + ((u >> 16) & 1u);   // round-to-nearest-even
    return (unsigned short)(r >> 16);
}
static __device__ __forceinline__ float bf2f(unsigned short b) {
    return __uint_as_float(((unsigned)b) << 16);
}

// ---------------------------------------------------------------------------
__global__ __launch_bounds__(256) void cvt_bf16_kernel(
    const float* __restrict__ in, unsigned short* __restrict__ out, size_t n) {
    size_t i = (size_t)blockIdx.x * blockDim.x + threadIdx.x;
    size_t stride = (size_t)gridDim.x * blockDim.x;
    for (; i < n; i += stride) out[i] = f2bf(in[i]);
}

// ---------------------------------------------------------------------------
// Gates: one wave32 per row. K=1024 strided over lanes, shfl reduce, softmax.
__global__ __launch_bounds__(256) void ple_gate_kernel(
    const float* __restrict__ x,      // [T][B][I]
    const float* __restrict__ gW,     // [T][I][6]
    const float* __restrict__ gb,     // [T][6]
    float* __restrict__ g)            // [T][B][6]
{
    int gwave = ((int)blockIdx.x * (int)blockDim.x + (int)threadIdx.x) >> 5;
    int lane  = threadIdx.x & 31;
    if (gwave >= NT * NB) return;
    int t = gwave / NB, row = gwave % NB;
    const float* xr = x + ((size_t)t * NB + row) * NI;
    const float* w  = gW + (size_t)t * NI * NE;
    float acc[NE] = {0.f, 0.f, 0.f, 0.f, 0.f, 0.f};
    for (int k = lane; k < NI; k += 32) {
        float xv = xr[k];
        const float* wk = w + k * NE;
        #pragma unroll
        for (int e = 0; e < NE; ++e) acc[e] += xv * wk[e];
    }
    #pragma unroll
    for (int e = 0; e < NE; ++e)
        #pragma unroll
        for (int off = 16; off > 0; off >>= 1)
            acc[e] += __shfl_xor(acc[e], off, 32);
    if (lane == 0) {
        #pragma unroll
        for (int e = 0; e < NE; ++e) acc[e] += gb[t * NE + e];
        float m = acc[0];
        #pragma unroll
        for (int e = 1; e < NE; ++e) m = fmaxf(m, acc[e]);
        float s = 0.f;
        #pragma unroll
        for (int e = 0; e < NE; ++e) { acc[e] = __expf(acc[e] - m); s += acc[e]; }
        float inv = 1.f / s;
        float* go = g + ((size_t)t * NB + row) * NE;
        #pragma unroll
        for (int e = 0; e < NE; ++e) go[e] = acc[e] * inv;
    }
}

// --------------------------- fragment helpers ------------------------------
// 16-bit A (16x32) / B (32x16) fragment per ISA layout:
// lanes 0-15 hold M/N = lane with K {half*8 .. +8} in VGPR0-3 and
// K {16+half*8 .. +8} in VGPR4-7  (half = lane>>4).  Two b128 LDS loads.
static __device__ __forceinline__ v16bf load_frag(const unsigned short* p) {
    v8bf lo = *(const v8bf*)p;          // K block 0: 8 bf16, 16 bytes
    v8bf hi = *(const v8bf*)(p + 16);   // K block 1 (+16 elements)
    return __builtin_shufflevector(lo, hi, 0, 1, 2, 3, 4, 5, 6, 7,
                                           8, 9, 10, 11, 12, 13, 14, 15);
}

// ---------------------------------------------------------------------------
// Pass C: shared experts.  grid = (NO/BN, NB/BM, 4).  S[e][b][o] bf16.
__global__ __launch_bounds__(256) void ple_shared_kernel(
    const unsigned short* __restrict__ xb0,   // [B][I] bf16 (task 0 input)
    const unsigned short* __restrict__ swb,   // [4][I][O] bf16
    const float* __restrict__ sb,             // [4][O]
    unsigned short* __restrict__ S)           // [4][B][O] bf16
{
    const int e = blockIdx.z;
    const int rowBase = blockIdx.y * BM;
    const int colBase = blockIdx.x * BN;
    const int tid = threadIdx.x;
    const int lane = tid & 31, wave = tid >> 5;
    const int waveM = wave >> 1, waveN = wave & 1;
    const int half = lane >> 4, l16 = lane & 15;

    __shared__ __align__(16) unsigned short shA[BM][KBP];
    __shared__ __align__(16) unsigned short shB[BN][KBP];
    __shared__ float shBias[BN];

    if (tid < BN) shBias[tid] = sb[(size_t)e * NO + colBase + tid];

    const unsigned short* xA = xb0 + (size_t)rowBase * NI;
    const unsigned short* wB = swb + (size_t)e * NI * NO;

    v8f acc[2][4] = {};

    for (int k0 = 0; k0 < NI; k0 += KB) {
        __syncthreads();
        {   // A tile: 128x64, row-major, b128 copies
            int r = tid >> 3, c = (tid & 7) * 8;
            #pragma unroll
            for (int p = 0; p < 4; ++p) {
                int rr = p * 32 + r;
                us8 v = *(const us8*)(xA + (size_t)rr * NI + k0 + c);
                *(us8*)(&shA[rr][c]) = v;
            }
        }
        {   // B tile: 64x128, transposed into [n][k]; 2 bf16 packed per b32
            int kp = (tid >> 4) * 2;     // k-pair base in 32-block
            int nc = (tid & 15) * 8;     // column group (coalesced global)
            #pragma unroll
            for (int p = 0; p < 2; ++p) {
                int k = k0 + p * 32 + kp;
                U8 w0, w1;
                w0.v = *(const us8*)(wB + (size_t)k * NO + colBase + nc);
                w1.v = *(const us8*)(wB + (size_t)(k + 1) * NO + colBase + nc);
                #pragma unroll
                for (int i = 0; i < 8; ++i) {
                    int ii = (i + tid) & 7;   // stagger -> spread LDS banks
                    unsigned pk = (unsigned)w0.s[ii] | ((unsigned)w1.s[ii] << 16);
                    *(unsigned*)&shB[nc + ii][p * 32 + kp] = pk;
                }
            }
        }
        __syncthreads();
        #pragma unroll
        for (int kk = 0; kk < KB; kk += 32) {
            v16bf a[2];
            #pragma unroll
            for (int mi = 0; mi < 2; ++mi)
                a[mi] = load_frag(&shA[waveM * 32 + mi * 16 + l16][kk + half * 8]);
            #pragma unroll
            for (int ni = 0; ni < 4; ++ni) {
                v16bf b = load_frag(&shB[waveN * 64 + ni * 16 + l16][kk + half * 8]);
                #pragma unroll
                for (int mi = 0; mi < 2; ++mi)
                    acc[mi][ni] = __builtin_amdgcn_wmma_f32_16x16x32_bf16(
                        false, a[mi], false, b, (short)0, acc[mi][ni], false, false);
            }
        }
    }

    // Epilogue: bias + relu, store bf16.  C/D layout: lane n=l16, rows half*8+v.
    #pragma unroll
    for (int mi = 0; mi < 2; ++mi)
        #pragma unroll
        for (int ni = 0; ni < 4; ++ni) {
            int colLoc = waveN * 64 + ni * 16 + l16;
            size_t col = colBase + colLoc;
            #pragma unroll
            for (int v = 0; v < 8; ++v) {
                size_t row = rowBase + waveM * 32 + mi * 16 + half * 8 + v;
                float r = acc[mi][ni][v] + shBias[colLoc];
                r = r > 0.f ? r : 0.f;
                S[((size_t)e * NB + row) * NO + col] = f2bf(r);
            }
        }
}

// ---------------------------------------------------------------------------
// Pass D: per-task experts + gated combine.  grid = (NO/BN, NB/BM, NT).
__global__ __launch_bounds__(256) void ple_task_kernel(
    const unsigned short* __restrict__ xb,   // [T][B][I] bf16
    const unsigned short* __restrict__ twb,  // [T][2][I][O] bf16
    const float* __restrict__ tb,            // [T][2][O]
    const unsigned short* __restrict__ S,    // [4][B][O] bf16
    const float* __restrict__ g,             // [T][B][6]
    float* __restrict__ out)                 // [T][B][O]
{
    const int t = blockIdx.z;
    const int rowBase = blockIdx.y * BM;
    const int colBase = blockIdx.x * BN;
    const int tid = threadIdx.x;
    const int lane = tid & 31, wave = tid >> 5;
    const int waveM = wave >> 1, waveN = wave & 1;
    const int half = lane >> 4, l16 = lane & 15;

    __shared__ __align__(16) unsigned short shA[BM][KBP];
    __shared__ __align__(16) unsigned short shB[2][BN][KBP];
    __shared__ float shBias[2][BN];
    __shared__ float shG[BM][8];

    if (tid < BM) {
        const float* gp = g + ((size_t)t * NB + rowBase + tid) * NE;
        #pragma unroll
        for (int e = 0; e < NE; ++e) shG[tid][e] = gp[e];
    }
    {
        int e = tid >> 7, c = tid & 127;
        shBias[e][c] = tb[((size_t)t * 2 + e) * NO + colBase + c];
    }

    const unsigned short* xA = xb + ((size_t)t * NB + rowBase) * NI;

    v8f acc[2][2][4] = {};   // [expert][mi][ni]

    for (int k0 = 0; k0 < NI; k0 += KB) {
        __syncthreads();
        {   // A tile
            int r = tid >> 3, c = (tid & 7) * 8;
            #pragma unroll
            for (int p = 0; p < 4; ++p) {
                int rr = p * 32 + r;
                us8 v = *(const us8*)(xA + (size_t)rr * NI + k0 + c);
                *(us8*)(&shA[rr][c]) = v;
            }
        }
        {   // B tiles for both task experts, transposed, b32-packed stores
            int kp = (tid >> 4) * 2;
            int nc = (tid & 15) * 8;
            #pragma unroll
            for (int e = 0; e < 2; ++e) {
                const unsigned short* wB = twb + ((size_t)t * 2 + e) * NI * NO;
                #pragma unroll
                for (int p = 0; p < 2; ++p) {
                    int k = k0 + p * 32 + kp;
                    U8 w0, w1;
                    w0.v = *(const us8*)(wB + (size_t)k * NO + colBase + nc);
                    w1.v = *(const us8*)(wB + (size_t)(k + 1) * NO + colBase + nc);
                    #pragma unroll
                    for (int i = 0; i < 8; ++i) {
                        int ii = (i + tid) & 7;
                        unsigned pk = (unsigned)w0.s[ii] | ((unsigned)w1.s[ii] << 16);
                        *(unsigned*)&shB[e][nc + ii][p * 32 + kp] = pk;
                    }
                }
            }
        }
        __syncthreads();
        #pragma unroll
        for (int kk = 0; kk < KB; kk += 32) {
            v16bf a[2];
            #pragma unroll
            for (int mi = 0; mi < 2; ++mi)
                a[mi] = load_frag(&shA[waveM * 32 + mi * 16 + l16][kk + half * 8]);
            #pragma unroll
            for (int e = 0; e < 2; ++e)
                #pragma unroll
                for (int ni = 0; ni < 4; ++ni) {
                    v16bf b = load_frag(&shB[e][waveN * 64 + ni * 16 + l16][kk + half * 8]);
                    #pragma unroll
                    for (int mi = 0; mi < 2; ++mi)
                        acc[e][mi][ni] = __builtin_amdgcn_wmma_f32_16x16x32_bf16(
                            false, a[mi], false, b, (short)0, acc[e][mi][ni], false, false);
                }
        }
    }
    __syncthreads();

    // Epilogue: bias+relu, gated combine with 4 shared experts from S.
    #pragma unroll
    for (int mi = 0; mi < 2; ++mi)
        #pragma unroll
        for (int ni = 0; ni < 4; ++ni) {
            int colLoc = waveN * 64 + ni * 16 + l16;
            size_t col = colBase + colLoc;
            #pragma unroll
            for (int v = 0; v < 8; ++v) {
                int rowLoc = waveM * 32 + mi * 16 + half * 8 + v;
                size_t row = rowBase + rowLoc;
                float r0 = acc[0][mi][ni][v] + shBias[0][colLoc];
                float r1 = acc[1][mi][ni][v] + shBias[1][colLoc];
                r0 = r0 > 0.f ? r0 : 0.f;
                r1 = r1 > 0.f ? r1 : 0.f;
                float o = shG[rowLoc][4] * r0 + shG[rowLoc][5] * r1;
                #pragma unroll
                for (int e = 0; e < 4; ++e)
                    o += shG[rowLoc][e] *
                         bf2f(S[((size_t)e * NB + row) * NO + col]);
                out[((size_t)t * NB + row) * NO + col] = o;
            }
        }
}

// ---------------------------------------------------------------------------
extern "C" void kernel_launch(void* const* d_in, const int* in_sizes, int n_in,
                              void* d_out, int out_size, void* d_ws, size_t ws_size,
                              hipStream_t stream) {
    const float* task_inputs = (const float*)d_in[0]; // [3][16384][1024]
    const float* shared_W    = (const float*)d_in[1]; // [4][1024][512]
    const float* shared_b    = (const float*)d_in[2]; // [4][512]
    const float* task_W      = (const float*)d_in[3]; // [3][2][1024][512]
    const float* task_b      = (const float*)d_in[4]; // [3][2][512]
    const float* gate_W      = (const float*)d_in[5]; // [3][1024][6]
    const float* gate_b      = (const float*)d_in[6]; // [3][6]
    float* out = (float*)d_out;                        // [3][16384][512]

    // Workspace layout (~171 MB total)
    char* w = (char*)d_ws;
    unsigned short* xb  = (unsigned short*)w; w += (size_t)NT * NB * NI * 2;  // 100.7 MB
    unsigned short* swb = (unsigned short*)w; w += (size_t)4 * NI * NO * 2;   //   4.2 MB
    unsigned short* twb = (unsigned short*)w; w += (size_t)NT * 2 * NI * NO * 2; // 6.3 MB
    unsigned short* S   = (unsigned short*)w; w += (size_t)4 * NB * NO * 2;   //  67.1 MB
    float* g            = (float*)w;                                          //   1.2 MB

    // 1) fp32 -> bf16 conversions
    cvt_bf16_kernel<<<4096, 256, 0, stream>>>(task_inputs, xb, (size_t)NT * NB * NI);
    cvt_bf16_kernel<<<512, 256, 0, stream>>>(shared_W, swb, (size_t)4 * NI * NO);
    cvt_bf16_kernel<<<512, 256, 0, stream>>>(task_W, twb, (size_t)NT * 2 * NI * NO);

    // 2) gates (fp32 math, softmax over 6 experts)
    ple_gate_kernel<<<(NT * NB * 32) / 256, 256, 0, stream>>>(task_inputs, gate_W, gate_b, g);

    // 3) shared experts once (no per-task recompute): S = relu(x0.Ws + bs)
    dim3 gridS(NO / BN, NB / BM, 4);
    ple_shared_kernel<<<gridS, 256, 0, stream>>>(xb, swb, shared_b, S);

    // 4) per-task experts + gated combine
    dim3 gridT(NO / BN, NB / BM, NT);
    ple_task_kernel<<<gridT, 256, 0, stream>>>(xb, twb, task_b, S, g, out);
}
